// ODEBlock_54546084659475
// MI455X (gfx1250) — matrix operand order repaired and checked by reading, not compiled, not run on hardware
//
#include <hip/hip_runtime.h>

// ---------------------------------------------------------------------------
// Neural ODE block for MI455X (gfx1250, wave32, WMMA).
// dy/dt = W3*relu(W2*relu(W1*y+b1)+b2)+b3, integrated t: 0 -> 1 with RK4 (8 steps).
// All GEMM FLOPs go through v_wmma_f32_16x16x32_bf16 (bf16 in, f32 acc).
// Weights are pre-packed once per launch into the exact WMMA B-fragment
// layout so B fragments stream from L2 with coalesced b128 loads (no LDS
// transpose). A tiles are double-buffered in LDS (1 barrier / K-step).
// ---------------------------------------------------------------------------

typedef __attribute__((ext_vector_type(16))) __bf16 v16bf;
typedef __attribute__((ext_vector_type(8)))  __bf16 v8bf;
typedef __attribute__((ext_vector_type(8)))  float  v8f;

union FragAB {
    v16bf v;
    v8bf  h[2];
};

template <typename T> struct IsF32        { static constexpr bool v = false; };
template <>           struct IsF32<float> { static constexpr bool v = true;  };

// ---------------------------------------------------------------------------
// Pack fp32 weights W[K][N] (row-major) into WMMA B-fragment order, bf16:
//   Wp[ ((nt*(K/32) + kt)*32 + lane)*16 + e ]
//     lane 0..31: n = nt*16 + lane%16 ; K chunk = kt*32 + (lane/16)*16
//     e  0..15 : K offset within chunk
// One thread per (nt,kt,lane): writes 16 contiguous bf16 (32 B).
// ---------------------------------------------------------------------------
__global__ __launch_bounds__(256)
void pack_w_kernel(const float* __restrict__ W, __bf16* __restrict__ Wp,
                   int K, int N)
{
    int t = blockIdx.x * blockDim.x + threadIdx.x;
    const int total = (N >> 4) * (K >> 5) * 32;
    if (t >= total) return;
    const int lane = t & 31;
    const int ktnt = t >> 5;
    const int nkt  = K >> 5;
    const int nt   = ktnt / nkt;
    const int kt   = ktnt - nt * nkt;
    const int n    = nt * 16 + (lane & 15);
    const int kb   = kt * 32 + (lane >> 4) * 16;
    v8bf o0, o1;
#pragma unroll
    for (int e = 0; e < 8; ++e) {
        o0[e] = (__bf16)W[(size_t)(kb + e) * N + n];
        o1[e] = (__bf16)W[(size_t)(kb + 8 + e) * N + n];
    }
    v8bf* dst = reinterpret_cast<v8bf*>(Wp + (size_t)t * 16);
    dst[0] = o0;
    dst[1] = o1;
}

// ---------------------------------------------------------------------------
// Tiled bf16 WMMA GEMM:  C[M,N] = act(A[M,K] * B[K,N] + bias[N])
//   A  : row-major fp32 or bf16; staged to LDS as bf16, double-buffered.
//   Wp : weights pre-packed in B-fragment order (see pack_w_kernel).
//   Block tile 128(M) x 256(N) x 32(K); 256 threads = 8 waves (2Mx4N),
//   each wave owns a 64x64 tile = 4x4 accumulators -> 16 WMMA / K-step.
// ---------------------------------------------------------------------------
template <typename TIN, typename TOUT, bool RELU>
__global__ __launch_bounds__(256)
void gemm_bf16_wmma(const TIN* __restrict__ A,
                    const __bf16* __restrict__ Wp,
                    const float* __restrict__ bias,
                    TOUT* __restrict__ C,
                    int M, int N, int K)
{
    constexpr int LDA = 32;                       // As row stride (bf16)
    __shared__ __align__(16) __bf16 As[2][128 * LDA];   // 2 x 8 KB

    const int tid     = threadIdx.x;
    const int lane    = tid & 31;                 // wave32
    const int wid     = tid >> 5;                 // 0..7
    const int wm      = wid >> 2;                 // 0..1 -> 64 rows
    const int wn      = wid & 3;                  // 0..3 -> 64 cols
    const int lane_lo = lane & 15;
    const int khalf   = lane >> 4;

    const int bm  = blockIdx.y * 128;
    const int bn  = blockIdx.x * 256;
    const int nkt = K >> 5;                       // number of K-steps

    v8f acc[4][4];
#pragma unroll
    for (int mi = 0; mi < 4; ++mi)
#pragma unroll
        for (int ni = 0; ni < 4; ++ni) {
            v8f z = {};
            acc[mi][ni] = z;
        }

    // ---- A staging assignment: thread -> (row, 16-col half) -----------------
    const int arow = tid >> 1;                    // 0..127
    const int acol = (tid & 1) * 16;              // 0 or 16
    const TIN* aptr = A + (size_t)(bm + arow) * K + acol;

    auto loadA = [&](int kk, v8bf& o0, v8bf& o1) {
        const TIN* src = aptr + kk;
        if constexpr (IsF32<TIN>::v) {
            const float4* s4 = reinterpret_cast<const float4*>(src);
            float4 f0 = s4[0], f1 = s4[1], f2 = s4[2], f3 = s4[3];
            o0[0] = (__bf16)f0.x; o0[1] = (__bf16)f0.y;
            o0[2] = (__bf16)f0.z; o0[3] = (__bf16)f0.w;
            o0[4] = (__bf16)f1.x; o0[5] = (__bf16)f1.y;
            o0[6] = (__bf16)f1.z; o0[7] = (__bf16)f1.w;
            o1[0] = (__bf16)f2.x; o1[1] = (__bf16)f2.y;
            o1[2] = (__bf16)f2.z; o1[3] = (__bf16)f2.w;
            o1[4] = (__bf16)f3.x; o1[5] = (__bf16)f3.y;
            o1[6] = (__bf16)f3.z; o1[7] = (__bf16)f3.w;
        } else {
            const v8bf* s8 = reinterpret_cast<const v8bf*>(src);
            o0 = s8[0];
            o1 = s8[1];
        }
    };

    // ---- B fragment base pointers: one per ni, advance 512 elems per kt -----
    const __bf16* bp[4];
#pragma unroll
    for (int ni = 0; ni < 4; ++ni) {
        const int nt = (bn >> 4) + wn * 4 + ni;
        bp[ni] = Wp + ((size_t)nt * nkt * 32 + lane) * 16;
    }

    // ---- prologue: stage tile 0 --------------------------------------------
    {
        v8bf p0, p1;
        loadA(0, p0, p1);
        v8bf* dst = reinterpret_cast<v8bf*>(&As[0][arow * LDA + acol]);
        dst[0] = p0;
        dst[1] = p1;
    }
    __syncthreads();

    for (int kt = 0; kt < nkt; ++kt) {
        const int cur = kt & 1;

        // next A tile: global loads issued early, stored after compute
        v8bf p0, p1;
        const bool have_next = (kt + 1) < nkt;
        if (have_next) loadA((kt + 1) * 32, p0, p1);
        if (kt + 2 < nkt)
            __builtin_prefetch(aptr + (size_t)(kt + 2) * 32, 0, 1);

        // B fragments straight from L2 (coalesced 32 B / lane)
        FragAB bfr[4];
#pragma unroll
        for (int ni = 0; ni < 4; ++ni) {
            const v8bf* s = reinterpret_cast<const v8bf*>(bp[ni] + (size_t)kt * 512);
            bfr[ni].h[0] = s[0];
            bfr[ni].h[1] = s[1];
        }

        // A fragments from LDS (lane half h holds contiguous K = h*16..h*16+15, as packed B)
        FragAB afr[4];
#pragma unroll
        for (int mi = 0; mi < 4; ++mi) {
            const __bf16* p = &As[cur][(wm * 64 + mi * 16 + lane_lo) * LDA + khalf * 16];
            afr[mi].h[0] = *reinterpret_cast<const v8bf*>(p);       // K = h*16..h*16+7
            afr[mi].h[1] = *reinterpret_cast<const v8bf*>(p + 8);   // K = h*16+8..h*16+15
        }

#pragma unroll
        for (int mi = 0; mi < 4; ++mi)
#pragma unroll
            for (int ni = 0; ni < 4; ++ni)
                acc[mi][ni] = __builtin_amdgcn_wmma_f32_16x16x32_bf16(
                    false, afr[mi].v, false, bfr[ni].v,
                    (short)0, acc[mi][ni], false, false);

        if (have_next) {
            v8bf* dst = reinterpret_cast<v8bf*>(&As[1 - cur][arow * LDA + acol]);
            dst[0] = p0;
            dst[1] = p1;
        }
        __syncthreads();
    }

    // ---- epilogue: bias (+ReLU); 32-bit offsets off one 64-bit base ---------
    float bv[4];
#pragma unroll
    for (int ni = 0; ni < 4; ++ni)
        bv[ni] = bias[bn + wn * 64 + ni * 16 + lane_lo];

    TOUT* cbase = C + (size_t)(bm + wm * 64 + khalf * 8) * N
                    + (bn + wn * 64 + lane_lo);
#pragma unroll
    for (int mi = 0; mi < 4; ++mi)
#pragma unroll
        for (int ni = 0; ni < 4; ++ni)
#pragma unroll
            for (int r = 0; r < 8; ++r) {
                float v = acc[mi][ni][r] + bv[ni];
                if constexpr (RELU) v = fmaxf(v, 0.0f);
                cbase[(unsigned)((mi * 16 + r) * N + ni * 16)] = (TOUT)v;
            }
}

// ---------------------------------------------------------------------------
// RK4 combine kernels (float4-vectorized)
//   stage: acc = first ? wk*k : acc + wk*k ;  yt = y + a*k
//   final: y  += s * (acc + k)
// ---------------------------------------------------------------------------
__global__ __launch_bounds__(256)
void rk_stage_kernel(const float* __restrict__ y, const float* __restrict__ k,
                     float* __restrict__ acc, float* __restrict__ yt,
                     float wk, float a, int first, int n4)
{
    int i = blockIdx.x * blockDim.x + threadIdx.x;
    if (i >= n4) return;
    float4 kv = reinterpret_cast<const float4*>(k)[i];
    float4 yv = reinterpret_cast<const float4*>(y)[i];
    float4 av;
    if (first) {
        av.x = wk * kv.x; av.y = wk * kv.y; av.z = wk * kv.z; av.w = wk * kv.w;
    } else {
        float4 a0 = reinterpret_cast<const float4*>(acc)[i];
        av.x = a0.x + wk * kv.x; av.y = a0.y + wk * kv.y;
        av.z = a0.z + wk * kv.z; av.w = a0.w + wk * kv.w;
    }
    reinterpret_cast<float4*>(acc)[i] = av;
    float4 tv;
    tv.x = yv.x + a * kv.x; tv.y = yv.y + a * kv.y;
    tv.z = yv.z + a * kv.z; tv.w = yv.w + a * kv.w;
    reinterpret_cast<float4*>(yt)[i] = tv;
}

__global__ __launch_bounds__(256)
void rk_final_kernel(float* __restrict__ y, const float* __restrict__ acc,
                     const float* __restrict__ k, float s, int n4)
{
    int i = blockIdx.x * blockDim.x + threadIdx.x;
    if (i >= n4) return;
    float4 yv = reinterpret_cast<const float4*>(y)[i];
    float4 av = reinterpret_cast<const float4*>(acc)[i];
    float4 kv = reinterpret_cast<const float4*>(k)[i];
    yv.x += s * (av.x + kv.x); yv.y += s * (av.y + kv.y);
    yv.z += s * (av.z + kv.z); yv.w += s * (av.w + kv.w);
    reinterpret_cast<float4*>(y)[i] = yv;
}

// ---------------------------------------------------------------------------
// Host-side launcher
// ---------------------------------------------------------------------------
extern "C" void kernel_launch(void* const* d_in, const int* in_sizes, int n_in,
                              void* d_out, int out_size, void* d_ws, size_t ws_size,
                              hipStream_t stream)
{
    constexpr int NB = 8192;   // batch
    constexpr int ND = 1024;   // data dim
    constexpr int NH = 2048;   // hidden dim
    constexpr int NSTEPS = 8;  // RK4 steps over t in [0,1]

    const float* x  = (const float*)d_in[0];
    const float* W1 = (const float*)d_in[1];
    const float* b1 = (const float*)d_in[2];
    const float* W2 = (const float*)d_in[3];
    const float* b2 = (const float*)d_in[4];
    const float* W3 = (const float*)d_in[5];
    const float* b3 = (const float*)d_in[6];
    float* y = (float*)d_out;

    char* ws = (char*)d_ws;
    __bf16* W1p = (__bf16*)ws; ws += (size_t)ND * NH * sizeof(__bf16);
    __bf16* W2p = (__bf16*)ws; ws += (size_t)NH * NH * sizeof(__bf16);
    __bf16* W3p = (__bf16*)ws; ws += (size_t)NH * ND * sizeof(__bf16);
    __bf16* H1b = (__bf16*)ws; ws += (size_t)NB * NH * sizeof(__bf16);
    __bf16* H2b = (__bf16*)ws; ws += (size_t)NB * NH * sizeof(__bf16);
    float*  F   = (float*)ws;  ws += (size_t)NB * ND * sizeof(float);
    float*  ACC = (float*)ws;  ws += (size_t)NB * ND * sizeof(float);
    float*  YT  = (float*)ws;  ws += (size_t)NB * ND * sizeof(float);

    // --- pack weights to bf16 B-fragment layout (once per launch) ------------
    {
        int nt;
        nt = (ND * NH) / 16;    // (N/16)*(K/32)*32 threads
        pack_w_kernel<<<(nt + 255) / 256, 256, 0, stream>>>(W1, W1p, ND, NH);
        nt = (NH * NH) / 16;
        pack_w_kernel<<<(nt + 255) / 256, 256, 0, stream>>>(W2, W2p, NH, NH);
        nt = (NH * ND) / 16;
        pack_w_kernel<<<(nt + 255) / 256, 256, 0, stream>>>(W3, W3p, NH, ND);
    }

    // --- y = x ---------------------------------------------------------------
    hipMemcpyAsync(y, x, (size_t)NB * ND * sizeof(float),
                   hipMemcpyDeviceToDevice, stream);

    const dim3 blk(256);
    const dim3 g1(NH / 256, NB / 128);   // GEMM1: [NB,ND]x[ND,NH]
    const dim3 g2(NH / 256, NB / 128);   // GEMM2: [NB,NH]x[NH,NH]
    const dim3 g3(ND / 256, NB / 128);   // GEMM3: [NB,NH]x[NH,ND]

    auto feval = [&](const float* yin) {
        gemm_bf16_wmma<float,  __bf16, true ><<<g1, blk, 0, stream>>>(yin, W1p, b1, H1b, NB, NH, ND);
        gemm_bf16_wmma<__bf16, __bf16, true ><<<g2, blk, 0, stream>>>(H1b, W2p, b2, H2b, NB, NH, NH);
        gemm_bf16_wmma<__bf16, float,  false><<<g3, blk, 0, stream>>>(H2b, W3p, b3, F,   NB, ND, NH);
    };

    const int   n4 = (NB * ND) / 4;
    const int   ng = (n4 + 255) / 256;
    const float h  = 1.0f / (float)NSTEPS;

    for (int s = 0; s < NSTEPS; ++s) {
        feval(y);                                                              // k1
        rk_stage_kernel<<<ng, blk, 0, stream>>>(y, F, ACC, YT, 1.0f, 0.5f * h, 1, n4);
        feval(YT);                                                             // k2
        rk_stage_kernel<<<ng, blk, 0, stream>>>(y, F, ACC, YT, 2.0f, 0.5f * h, 0, n4);
        feval(YT);                                                             // k3
        rk_stage_kernel<<<ng, blk, 0, stream>>>(y, F, ACC, YT, 2.0f, h,        0, n4);
        feval(YT);                                                             // k4
        rk_final_kernel<<<ng, blk, 0, stream>>>(y, ACC, F, h / 6.0f, n4);
    }

    (void)in_sizes; (void)n_in; (void)out_size; (void)ws_size;
}
